// Simp_Model_LSTM_54872502174221
// MI455X (gfx1250) — matrix operand order repaired and checked by reading, not compiled, not run on hardware
//
#include <hip/hip_runtime.h>
#include <hip/hip_bf16.h>

// ---------------------------------------------------------------------------
// Fused 3-layer LSTM towers (q: T=12, p: T=50) + head, bf16 WMMA on gfx1250.
// One wave (32 lanes) per block; each block owns 16 batch rows (WMMA M=16).
// GEMMs: v_wmma_f32_16x16x32_bf16 (f32 accumulate). x staged to LDS with
// double-buffered GLOBAL_LOAD_ASYNC_TO_LDS_B128 (ASYNCcnt) when available.
// ---------------------------------------------------------------------------

typedef __attribute__((ext_vector_type(16))) __bf16 bf16x16;
typedef __attribute__((ext_vector_type(8)))  float  f32x8;
typedef __attribute__((ext_vector_type(4)))  int    v4i;

#define BT 16      // batch tile (WMMA M)
#define CH 10      // time-chunk length
#define XBUFELEMS (BT * CH * 50)

#if defined(__HIP_DEVICE_COMPILE__) && __has_builtin(__builtin_amdgcn_global_load_async_to_lds_b128)
#define HAVE_ASYNC_LDS 1
#else
#define HAVE_ASYNC_LDS 0
#endif

// Padded layer geometry (K padded to 32, gates padded to 16):
//  L0: G=100 GP=112 Din=50 (KX=64, 2 chunks)  H=25 (KH=32)
//  L1: G=40  GP=48  Din=25 (KX=32, 1 chunk)   H=10
//  L2: G=16  GP=16  Din=10 (KX=32, 1 chunk)   H=4
//
// ws blob per tower (30400 B):
//  bf16: Wx0[112*64]@0  Wh0[112*32]@7168  Wx1[48*32]@10752 Wh1[48*32]@12288
//        Wx2[16*32]@13824 Wh2[16*32]@14336   (14848 bf16 = 29696 B)
//  f32 : bias0[112] bias1[48] bias2[16] @ byte 29696 (704 B)
#define TOWER_BYTES 30400
#define BIAS_BYTE_OFF 29696

// ---------------------------- prep kernel ---------------------------------
__global__ void prep_weights(const float* __restrict__ Wih0, const float* __restrict__ Whh0, const float* __restrict__ b0,
                             const float* __restrict__ Wih1, const float* __restrict__ Whh1, const float* __restrict__ b1,
                             const float* __restrict__ Wih2, const float* __restrict__ Whh2, const float* __restrict__ b2,
                             unsigned char* __restrict__ dst)
{
    __bf16* w   = (__bf16*)dst;
    float* bias = (float*)(dst + BIAS_BYTE_OFF);
    int tid    = blockIdx.x * blockDim.x + threadIdx.x;
    int stride = gridDim.x * blockDim.x;

    for (int i = tid; i < 112*64; i += stride) { int g = i >> 6, k = i & 63;
        w[i]         = (__bf16)((g < 100 && k < 50) ? Wih0[g*50 + k] : 0.f); }
    for (int i = tid; i < 112*32; i += stride) { int g = i >> 5, k = i & 31;
        w[7168 + i]  = (__bf16)((g < 100 && k < 25) ? Whh0[g*25 + k] : 0.f); }
    for (int i = tid; i < 48*32; i += stride)  { int g = i >> 5, k = i & 31;
        w[10752 + i] = (__bf16)((g < 40 && k < 25) ? Wih1[g*25 + k] : 0.f); }
    for (int i = tid; i < 48*32; i += stride)  { int g = i >> 5, k = i & 31;
        w[12288 + i] = (__bf16)((g < 40 && k < 10) ? Whh1[g*10 + k] : 0.f); }
    for (int i = tid; i < 16*32; i += stride)  { int g = i >> 5, k = i & 31;
        w[13824 + i] = (__bf16)((g < 16 && k < 10) ? Wih2[g*10 + k] : 0.f); }
    for (int i = tid; i < 16*32; i += stride)  { int g = i >> 5, k = i & 31;
        w[14336 + i] = (__bf16)((g < 16 && k < 4)  ? Whh2[g*4  + k] : 0.f); }
    for (int i = tid; i < 112; i += stride) bias[i]       = (i < 100) ? b0[i] : 0.f;
    for (int i = tid; i < 48;  i += stride) bias[112 + i] = (i < 40)  ? b1[i] : 0.f;
    for (int i = tid; i < 16;  i += stride) bias[160 + i] = b2[i];
}

// ------------------------- async copy primitives ---------------------------
__device__ __forceinline__ void copy16_async(const void* gsrc, void* ldst)
{
#if HAVE_ASYNC_LDS
    void* g = const_cast<void*>(gsrc);
    __builtin_amdgcn_global_load_async_to_lds_b128(
        (__attribute__((address_space(1))) v4i*)g,
        (__attribute__((address_space(3))) v4i*)ldst, 0, 0);
#else
    *(uint4*)ldst = *(const uint4*)gsrc;
#endif
}

__device__ __forceinline__ void wait_async0()
{
#if HAVE_ASYNC_LDS
#if __has_builtin(__builtin_amdgcn_s_wait_asynccnt)
    __builtin_amdgcn_s_wait_asynccnt(0);
#else
    asm volatile("s_wait_asynccnt 0" ::: "memory");
#endif
    asm volatile("" ::: "memory");   // keep LDS reads below the wait
#endif
}

// stage one time-chunk of x (f32, raw bytes) into LDS; 16B-aligned b128 copies.
// LDS layout: [16 rows][CH*50 f32], only first S*50 of each row live.
__device__ __forceinline__ void stage_x_chunk(const float* __restrict__ xg, int b0, int T,
                                              int cs, int S, float* buf)
{
    int lane = threadIdx.x & 31;
    int nvec = (S * 50) >> 2;            // float4s per row (S*50 % 4 == 0)
    int tot  = BT * nvec;
    for (int i = lane; i < tot; i += 32) {
        int m = i / nvec, v = i - m * nvec;
        const float* src = xg + (((size_t)(b0 + m)) * T + cs) * 50 + v * 4;
        float* dst = buf + m * (CH * 50) + v * 4;
        copy16_async(src, dst);
    }
}

// ---------------------- WMMA fragment gathers (wave32) ---------------------
// A (16xK bf16): lane<16 holds row=lane&15, K={hi*8..+7, hi*8+16..+23}, hi=lane>>4.
__device__ __forceinline__ bf16x16 load_A(const __bf16* base, int stride)
{
    int lane = threadIdx.x & 31, n = lane & 15, hi = lane >> 4;
    const __bf16* p = base + n * stride + hi * 8;
    union { bf16x16 v; uint32_t u[8]; } r;
    const uint32_t* a = (const uint32_t*)p;
    const uint32_t* b = (const uint32_t*)(p + 16);
#pragma unroll
    for (int i = 0; i < 4; ++i) { r.u[i] = a[i]; r.u[4 + i] = b[i]; }
    return r.v;
}

// A-fragment from f32 LDS (layer-0 x), converting to bf16 and zero-padding
// features >= 50. kbase is 0 or 32 (compile-time foldable guards).
__device__ __forceinline__ bf16x16 load_A_f32pad(const float* base, int stride, int kbase)
{
    int lane = threadIdx.x & 31, n = lane & 15, hi = lane >> 4;
    const float* p = base + n * stride + kbase + hi * 8;
    bf16x16 r;
#pragma unroll
    for (int i = 0; i < 8; ++i) {
        int k0 = kbase + hi * 8 + i;
        int k1 = k0 + 16;
        float v0 = (k0 < 50) ? p[i]      : 0.f;
        float v1 = (k1 < 50) ? p[i + 16] : 0.f;
        r[i]     = (__bf16)v0;
        r[8 + i] = (__bf16)v1;
    }
    return r;
}

// B (32x16 bf16, B[k][n] = W[row+n][k]): lane holds column n=lane&15,
// K = koff + {hi*8..+7, hi*8+16..+23}.
__device__ __forceinline__ bf16x16 load_B(const __bf16* wbase, int row, int stride, int koff)
{
    int lane = threadIdx.x & 31, n = lane & 15, hi = lane >> 4;
    const uint32_t* p = (const uint32_t*)(wbase + (size_t)(row + n) * stride + koff + hi * 8);
    union { bf16x16 v; uint32_t u[8]; } r;
#pragma unroll
    for (int i = 0; i < 4; ++i) { r.u[i] = p[i]; r.u[4 + i] = p[8 + i]; }
    return r.v;
}

// ------------------------------ LSTM step ----------------------------------
__device__ void lstm_step(bf16x16 ax0, bf16x16 ax1, int kxch,
                          const __bf16* Wx, int wxstride,
                          const __bf16* Wh,
                          const float* bias, int GP, int H,
                          __bf16* hprev, float* cbuf, float* gates,
                          __bf16* hseq_out, int seqstride, float* hf32_out)
{
    int lane = threadIdx.x & 31, n = lane & 15, hi = lane >> 4;

    bf16x16 ah = load_A(hprev, 32);

    int nt = GP >> 4;
    for (int j = 0; j < nt; ++j) {
        f32x8 acc;
        float bv = bias[j * 16 + n];
#pragma unroll
        for (int r = 0; r < 8; ++r) acc[r] = bv;

        acc = __builtin_amdgcn_wmma_f32_16x16x32_bf16(
                  false, ax0, false, load_B(Wx, j * 16, wxstride, 0),
                  (short)0, acc, false, false);
        if (kxch > 1)
            acc = __builtin_amdgcn_wmma_f32_16x16x32_bf16(
                      false, ax1, false, load_B(Wx, j * 16, wxstride, 32),
                      (short)0, acc, false, false);
        acc = __builtin_amdgcn_wmma_f32_16x16x32_bf16(
                  false, ah, false, load_B(Wh, j * 16, 32, 0),
                  (short)0, acc, false, false);

#pragma unroll
        for (int r = 0; r < 8; ++r) {
            int m = r + hi * 8;                    // C layout: M=r(+8), N=lane&15
            gates[m * GP + j * 16 + n] = acc[r];
        }
    }

    // pointwise LSTM cell update (gate order i,f,g,o)
    int total = BT * H;
    for (int idx = lane; idx < total; idx += 32) {
        int m = idx & 15, h0 = idx >> 4;
        float gi = gates[m * GP + h0];
        float gf = gates[m * GP + H + h0];
        float gg = gates[m * GP + 2 * H + h0];
        float go = gates[m * GP + 3 * H + h0];
        float c  = cbuf[m * 32 + h0];
        float si = 1.f / (1.f + __expf(-gi));
        float sf = 1.f / (1.f + __expf(-gf));
        float so = 1.f / (1.f + __expf(-go));
        c = sf * c + si * tanhf(gg);
        float h = so * tanhf(c);
        cbuf[m * 32 + h0]  = c;
        hprev[m * 32 + h0] = (__bf16)h;
        if (hseq_out) hseq_out[m * seqstride + h0] = (__bf16)h;
        if (hf32_out) hf32_out[m * 4 + h0] = h;
    }
}

// ------------------------------ main kernel --------------------------------
__global__ void __launch_bounds__(32)
Simp_Model_LSTM_54872502174221_kernel(const float* __restrict__ qin, const float* __restrict__ pin,
                                      const unsigned char* __restrict__ wsblob,
                                      const float* __restrict__ fW, const float* __restrict__ fb,
                                      float* __restrict__ out)
{
    __shared__ __align__(16) unsigned char sWblob[TOWER_BYTES];   // active tower weights
    __shared__ __align__(16) float  sXf32[2 * XBUFELEMS];         // double-buffered x chunks
    __shared__ __align__(16) __bf16 sSeq0[BT * CH * 32];          // layer0 output chunk
    __shared__ __align__(16) __bf16 sSeq1[BT * CH * 32];          // layer1 output chunk
    __shared__ __align__(16) float  sGates[BT * 112];
    __shared__ __align__(16) float  sC[3 * BT * 32];
    __shared__ __align__(16) __bf16 sHprev[3 * BT * 32];
    __shared__ __align__(16) float  sHq[BT * 4];
    __shared__ __align__(16) float  sHp[BT * 4];

    int lane = threadIdx.x & 31;
    int b0   = blockIdx.x * BT;

    // zero pad-sensitive buffers once (live parts are always re-written)
    { uint32_t* z = (uint32_t*)sSeq0; for (int i = lane; i < BT*CH*32/2; i += 32) z[i] = 0; }
    { uint32_t* z = (uint32_t*)sSeq1; for (int i = lane; i < BT*CH*32/2; i += 32) z[i] = 0; }

    for (int tower = 0; tower < 2; ++tower) {
        const float* xg = tower ? pin : qin;
        const int T     = tower ? 50 : 12;
        float* hfinal   = tower ? sHp : sHq;
        const int nch   = (T + CH - 1) / CH;

        // issue async staging: this tower's weights + first x chunk
        {
            const unsigned char* src = wsblob + (size_t)tower * TOWER_BYTES;
            for (int i = lane; i < TOWER_BYTES / 16; i += 32)
                copy16_async(src + i * 16, sWblob + i * 16);
        }
        stage_x_chunk(xg, b0, T, 0, min(CH, T), sXf32);

        const __bf16* w    = (const __bf16*)sWblob;
        const float*  bias = (const float*)(sWblob + BIAS_BYTE_OFF);
        const __bf16 *Wx0 = w,         *Wh0 = w + 7168;
        const __bf16 *Wx1 = w + 10752, *Wh1 = w + 12288;
        const __bf16 *Wx2 = w + 13824, *Wh2 = w + 14336;
        const float *bp0 = bias, *bp1 = bias + 112, *bp2 = bias + 160;

        // zero recurrent state (c, h) for all 3 layers
        { uint32_t* z = (uint32_t*)sC;     for (int i = lane; i < 3*BT*32;   i += 32) z[i] = 0; }
        { uint32_t* z = (uint32_t*)sHprev; for (int i = lane; i < 3*BT*32/2; i += 32) z[i] = 0; }

        for (int c = 0; c < nch; ++c) {
            wait_async0();                        // current buffer + weights ready
            const int cs = c * CH;
            const int S  = min(CH, T - cs);

            // overlap: start streaming the next chunk into the other buffer
            if (c + 1 < nch)
                stage_x_chunk(xg, b0, T, cs + CH, min(CH, T - cs - CH),
                              sXf32 + ((c + 1) & 1) * XBUFELEMS);

            const float* cur = sXf32 + (c & 1) * XBUFELEMS;

            // chunk-wise layer pipeline
            for (int tt = 0; tt < S; ++tt) {
                const float* xt = cur + tt * 50;
                bf16x16 a0 = load_A_f32pad(xt, CH * 50, 0);
                bf16x16 a1 = load_A_f32pad(xt, CH * 50, 32);
                lstm_step(a0, a1, 2, Wx0, 64, Wh0, bp0, 112, 25,
                          sHprev, sC, sGates, sSeq0 + tt * 32, CH * 32, nullptr);
            }
            for (int tt = 0; tt < S; ++tt) {
                bf16x16 a0 = load_A(sSeq0 + tt * 32, CH * 32);
                lstm_step(a0, a0, 1, Wx1, 32, Wh1, bp1, 48, 10,
                          sHprev + BT * 32, sC + BT * 32, sGates,
                          sSeq1 + tt * 32, CH * 32, nullptr);
            }
            for (int tt = 0; tt < S; ++tt) {
                bf16x16 a0 = load_A(sSeq1 + tt * 32, CH * 32);
                lstm_step(a0, a0, 1, Wx2, 32, Wh2, bp2, 16, 4,
                          sHprev + 2 * BT * 32, sC + 2 * BT * 32, sGates,
                          nullptr, 0, hfinal);
            }
        }
    }

    // head: scores = h_q * h_p ; softmax(scores @ fW^T + fb)
    if (lane < BT) {
        int m = lane;
        float s0 = sHq[m*4+0] * sHp[m*4+0];
        float s1 = sHq[m*4+1] * sHp[m*4+1];
        float s2 = sHq[m*4+2] * sHp[m*4+2];
        float s3 = sHq[m*4+3] * sHp[m*4+3];
        float l0 = fb[0] + s0*fW[0] + s1*fW[1] + s2*fW[2] + s3*fW[3];
        float l1 = fb[1] + s0*fW[4] + s1*fW[5] + s2*fW[6] + s3*fW[7];
        float mx = fmaxf(l0, l1);
        float e0 = __expf(l0 - mx), e1 = __expf(l1 - mx);
        float inv = 1.f / (e0 + e1);
        out[(size_t)(b0 + m) * 2 + 0] = e0 * inv;
        out[(size_t)(b0 + m) * 2 + 1] = e1 * inv;
    }
}

// ------------------------------ launcher -----------------------------------
extern "C" void kernel_launch(void* const* d_in, const int* in_sizes, int n_in,
                              void* d_out, int out_size, void* d_ws, size_t ws_size,
                              hipStream_t stream)
{
    (void)in_sizes; (void)n_in; (void)out_size; (void)ws_size;

    const float* q = (const float*)d_in[0];
    const float* p = (const float*)d_in[1];
    unsigned char* ws = (unsigned char*)d_ws;

    // tower q: inputs 2..10
    prep_weights<<<4, 256, 0, stream>>>(
        (const float*)d_in[2], (const float*)d_in[3], (const float*)d_in[4],
        (const float*)d_in[5], (const float*)d_in[6], (const float*)d_in[7],
        (const float*)d_in[8], (const float*)d_in[9], (const float*)d_in[10],
        ws);
    // tower p: inputs 11..19
    prep_weights<<<4, 256, 0, stream>>>(
        (const float*)d_in[11], (const float*)d_in[12], (const float*)d_in[13],
        (const float*)d_in[14], (const float*)d_in[15], (const float*)d_in[16],
        (const float*)d_in[17], (const float*)d_in[18], (const float*)d_in[19],
        ws + TOWER_BYTES);

    const float* fW = (const float*)d_in[20];
    const float* fb = (const float*)d_in[21];

    Simp_Model_LSTM_54872502174221_kernel<<<32768 / BT, 32, 0, stream>>>(
        q, p, ws, fW, fb, (float*)d_out);
}
